// GraphAutoEncoder_66305705116124
// MI455X (gfx1250) — matrix operand order repaired and checked by reading, not compiled, not run on hardware
//
#include <hip/hip_runtime.h>
#include <hip/hip_bf16.h>
#include <stdint.h>

typedef __attribute__((ext_vector_type(16))) _Float16     v16h;
typedef __attribute__((ext_vector_type(8)))  float        v8f;
typedef __attribute__((ext_vector_type(8)))  unsigned int v8u;

#define NN   16384      // nodes
#define NE   262144     // edges
#define NG   16         // graphs
#define CIN  128
#define CHID 256
#define CLAT 64

// Generic GEMM tile config: C[M,N] = A[M,K] * B[N,K]^T
#define BM 128
#define BN 64
#define BK 64
#define TN (BN / 16)
#define PAD 8

// ---------------------------------------------------------------------------
// Generic WMMA GEMM: C = act( A * B^T + bias ), A:[M,K] f32, B:[N,K] f32.
// f32 -> f16 staged in LDS, v_wmma_f32_16x16x32_f16, f32 accumulate.
// act: 0=none, 1=relu. (Used for the two small encoder GEMMs.)
// ---------------------------------------------------------------------------
__global__ __launch_bounds__(256) void k_gemm_abt(
    const float* __restrict__ A, const float* __restrict__ B,
    const float* __restrict__ bias, float* __restrict__ C,
    int M, int N, int K, int act)
{
    __shared__ _Float16 As[BM][BK + PAD];
    __shared__ _Float16 Bs[BN][BK + PAD];

    const int  tid  = threadIdx.x;
    const int  wave = tid >> 5;
    const int  lane = tid & 31;
    const int  l15  = lane & 15;
    const int  hsel = lane >> 4;
    const long m0   = (long)blockIdx.y * BM;
    const long n0   = (long)blockIdx.x * BN;

    v8f acc[TN] = {};

    for (int k0 = 0; k0 < K; k0 += BK) {
        for (int i = tid; i < BM * BK; i += 256) {
            int r = i >> 6, c = i & (BK - 1);
            As[r][c] = (_Float16)A[(m0 + r) * (long)K + (k0 + c)];
        }
        for (int i = tid; i < BN * BK; i += 256) {
            int r = i >> 6, c = i & (BK - 1);
            Bs[r][c] = (_Float16)B[(n0 + r) * (long)K + (k0 + c)];
        }
        __syncthreads();

        #pragma unroll
        for (int kk = 0; kk < BK; kk += 32) {
            v8u au;
            const int am = wave * 16 + l15;
            #pragma unroll
            for (int v = 0; v < 8; ++v) {
                const int k = kk + ((v >> 2) * 16) + (hsel * 8) + ((v & 3) * 2);
                au[v] = *reinterpret_cast<const unsigned int*>(&As[am][k]);
            }
            const v16h af = __builtin_bit_cast(v16h, au);

            #pragma unroll
            for (int t = 0; t < TN; ++t) {
                v8u bu;
                const int bn = t * 16 + l15;
                #pragma unroll
                for (int v = 0; v < 8; ++v) {
                    const int k = kk + ((v >> 2) * 16) + (hsel * 8) + ((v & 3) * 2);
                    bu[v] = *reinterpret_cast<const unsigned int*>(&Bs[bn][k]);
                }
                const v16h bf = __builtin_bit_cast(v16h, bu);
                acc[t] = __builtin_amdgcn_wmma_f32_16x16x32_f16(
                    false, af, false, bf, (short)0, acc[t], false, false);
            }
        }
        __syncthreads();
    }

    for (int t = 0; t < TN; ++t) {
        const long  gcol = n0 + t * 16 + l15;
        const float bv   = bias ? bias[gcol] : 0.0f;
        #pragma unroll
        for (int v = 0; v < 8; ++v) {
            const long grow = m0 + wave * 16 + hsel * 8 + v;
            float val = acc[t][v] + bv;
            if (act == 1) val = fmaxf(val, 0.0f);
            C[grow * (long)N + gcol] = val;
        }
    }
}

// ---------------------------------------------------------------------------
// a_hat = sigmoid(Z @ Z^T), Z pre-converted to f16 [NN x 64] (2 MB, L2-resident).
// Staging uses CDNA5 async copy: global_load_async_to_lds_b128 (ASYNCcnt),
// XOR-swizzled at 16B-chunk granularity to kill LDS bank conflicts.
// K = 64 -> single LDS stage, no K loop. NT stores for the 1.07 GB output.
// ---------------------------------------------------------------------------
__global__ __launch_bounds__(256) void k_ahat(const _Float16* __restrict__ zh,
                                              float* __restrict__ C)
{
    // A: rows 0..127 (128B/row), B: rows 128..191. 24 KB total, no padding
    // (async copies are raw), swizzle handles conflicts instead.
    __shared__ __align__(16) unsigned char smem[(128 + 64) * 128];

    const unsigned tid  = threadIdx.x;
    const unsigned wave = tid >> 5;
    const unsigned lane = tid & 31;
    const unsigned l15  = lane & 15;
    const unsigned hsel = lane >> 4;
    const long m0 = (long)blockIdx.y * 128;
    const long n0 = (long)blockIdx.x * 64;

    const unsigned sbase = (unsigned)(uintptr_t)&smem[0];  // LDS byte offset
    // 16B-chunk swizzled offset within smem (8 chunks per 128B row)
    auto loff = [](unsigned row, unsigned chunk) -> unsigned {
        return row * 128u + ((chunk ^ (row & 7u)) << 4);
    };

    // ---- async stage A: 1024 chunks (4 per thread) ----
    #pragma unroll
    for (int i = 0; i < 4; ++i) {
        const unsigned c = tid + i * 256;
        const unsigned row = c >> 3, ch = c & 7;
        const void* g = (const char*)zh + (m0 + row) * 128 + ch * 16;
        asm volatile("global_load_async_to_lds_b128 %0, %1, off"
                     :: "v"(sbase + loff(row, ch)), "v"(g) : "memory");
    }
    // ---- async stage B: 512 chunks (2 per thread) ----
    #pragma unroll
    for (int i = 0; i < 2; ++i) {
        const unsigned c = tid + i * 256;
        const unsigned row = c >> 3, ch = c & 7;
        const void* g = (const char*)zh + (n0 + row) * 128 + ch * 16;
        asm volatile("global_load_async_to_lds_b128 %0, %1, off"
                     :: "v"(sbase + loff(128 + row, ch)), "v"(g) : "memory");
    }
    asm volatile("s_wait_asynccnt 0" ::: "memory");
    __syncthreads();

    v8f acc[4] = {};
    const unsigned am = wave * 16 + l15;

    #pragma unroll
    for (int kk = 0; kk < 64; kk += 32) {
        const unsigned kc = kk >> 3;  // base 16B-chunk index (0 or 4)

        // A fragment: group g chunk = kc + 2g + hsel  [ISA 7.12.2 layout]
        const uint4 a0 = *(const uint4*)(smem + loff(am, kc + hsel));
        const uint4 a1 = *(const uint4*)(smem + loff(am, kc + 2 + hsel));
        v8u au;
        au[0] = a0.x; au[1] = a0.y; au[2] = a0.z; au[3] = a0.w;
        au[4] = a1.x; au[5] = a1.y; au[6] = a1.z; au[7] = a1.w;
        const v16h af = __builtin_bit_cast(v16h, au);

        #pragma unroll
        for (int t = 0; t < 4; ++t) {
            // B fragment: chunks kc + hsel, kc + 2 + hsel on row (n-tile col)
            const unsigned bn = 128 + t * 16 + l15;
            const uint4 b0 = *(const uint4*)(smem + loff(bn, kc + hsel));
            const uint4 b1 = *(const uint4*)(smem + loff(bn, kc + 2 + hsel));
            v8u bu;
            bu[0] = b0.x; bu[1] = b0.y; bu[2] = b0.z; bu[3] = b0.w;
            bu[4] = b1.x; bu[5] = b1.y; bu[6] = b1.z; bu[7] = b1.w;
            const v16h bf = __builtin_bit_cast(v16h, bu);
            acc[t] = __builtin_amdgcn_wmma_f32_16x16x32_f16(
                false, af, false, bf, (short)0, acc[t], false, false);
        }
    }

    // sigmoid epilogue + non-temporal stores (output never re-read)
    #pragma unroll
    for (int t = 0; t < 4; ++t) {
        const long gcol = n0 + t * 16 + l15;
        #pragma unroll
        for (int v = 0; v < 8; ++v) {
            const long grow = m0 + wave * 16 + hsel * 8 + v;
            const float val = 1.0f / (1.0f + __expf(-acc[t][v]));
            __builtin_nontemporal_store(val, C + grow * (long)NN + gcol);
        }
    }
}

// ---------------------------------------------------------------------------
// Elementwise / scatter helpers
// ---------------------------------------------------------------------------
__global__ void k_fill(float* __restrict__ p, float v, int n) {
    int i = blockIdx.x * 256 + threadIdx.x;
    if (i < n) p[i] = v;
}

__global__ void k_z2h(const float* __restrict__ z, _Float16* __restrict__ zh) {
    int i = blockIdx.x * 256 + threadIdx.x;          // NN*CLAT threads
    zh[i] = (_Float16)z[i];
}

__global__ void k_deg_edge(const int* __restrict__ ei, float* __restrict__ deg) {
    int e = blockIdx.x * 256 + threadIdx.x;
    if (e < NE) atomicAdd(&deg[ei[NE + e]], 1.0f);   // col = targets
}

__global__ void k_dinv(const float* __restrict__ deg, float* __restrict__ dinv) {
    int i = blockIdx.x * 256 + threadIdx.x;
    if (i < NN) dinv[i] = rsqrtf(deg[i]);            // deg >= 1 (self-loop)
}

__global__ void k_agg_init(const float* __restrict__ h, const float* __restrict__ dinv,
                           float* __restrict__ agg) {
    int i = blockIdx.x * 256 + threadIdx.x;          // NN*CHID threads
    int n = i >> 8;
    float d = dinv[n];
    agg[i] = h[i] * d * d;
}

__global__ __launch_bounds__(CHID) void k_agg_edge(
    const int* __restrict__ ei, const float* __restrict__ dinv,
    const float* __restrict__ h, float* __restrict__ agg) {
    const int e = blockIdx.x;
    const int c = threadIdx.x;
    const int r  = ei[e];        // source
    const int cl = ei[NE + e];   // target
    const float norm = dinv[r] * dinv[cl];
    atomicAdd(&agg[(long)cl * CHID + c], h[(long)r * CHID + c] * norm);
}

__global__ void k_bias_relu(float* __restrict__ agg, const float* __restrict__ b) {
    int i = blockIdx.x * 256 + threadIdx.x;          // NN*CHID threads
    agg[i] = fmaxf(agg[i] + b[i & (CHID - 1)], 0.0f);
}

__global__ void k_counts(const int* __restrict__ batch, float* __restrict__ counts) {
    int n = blockIdx.x * 256 + threadIdx.x;
    if (n < NN) atomicAdd(&counts[batch[n]], 1.0f);
}

__global__ void k_pool_sum(const int* __restrict__ batch, const float* __restrict__ z,
                           float* __restrict__ zsum) {
    int i = blockIdx.x * 256 + threadIdx.x;          // NN*CLAT threads
    int n = i >> 6;
    atomicAdd(&zsum[batch[n] * CLAT + (i & (CLAT - 1))], z[i]);
}

__global__ void k_pool_div(const float* __restrict__ zsum, const float* __restrict__ counts,
                           float* __restrict__ zg) {
    int i = blockIdx.x * 256 + threadIdx.x;
    if (i < NG * CLAT) zg[i] = zsum[i] / fmaxf(counts[i >> 6], 1.0f);
}

__global__ void k_decoder(const float* __restrict__ zg, const float* __restrict__ w,
                          const float* __restrict__ b, float* __restrict__ xg) {
    int i = blockIdx.x * 256 + threadIdx.x;
    if (i < NG * CIN) {
        int g = i >> 7, j = i & (CIN - 1);
        float s = b[j];
        #pragma unroll 8
        for (int k = 0; k < CLAT; ++k) s += zg[g * CLAT + k] * w[j * CLAT + k];
        xg[i] = s;
    }
}

__global__ void k_gather_xhat(const int* __restrict__ batch, const float* __restrict__ xg,
                              float* __restrict__ xhat) {
    int i = blockIdx.x * 256 + threadIdx.x;          // NN*CIN threads
    int n = i >> 7;
    xhat[i] = xg[batch[n] * CIN + (i & (CIN - 1))];
}

// ---------------------------------------------------------------------------
extern "C" void kernel_launch(void* const* d_in, const int* in_sizes, int n_in,
                              void* d_out, int out_size, void* d_ws, size_t ws_size,
                              hipStream_t stream) {
    const float* x     = (const float*)d_in[0];
    const int*   ei    = (const int*)  d_in[1];   // [2, NE] flat
    const int*   batch = (const int*)  d_in[2];
    const float* gcn_w = (const float*)d_in[3];
    const float* gcn_b = (const float*)d_in[4];
    const float* lin_w = (const float*)d_in[5];
    const float* lin_b = (const float*)d_in[6];
    const float* dec_w = (const float*)d_in[7];
    const float* dec_b = (const float*)d_in[8];

    float* out     = (float*)d_out;
    float* z_node  = out;                                  // NN*CLAT
    float* z_graph = out + (long)NN * CLAT;                // NG*CLAT
    float* x_hat   = z_graph + NG * CLAT;                  // NN*CIN
    float* a_hat   = x_hat + (long)NN * CIN;               // NN*NN

    float* ws     = (float*)d_ws;
    float* deg    = ws;                                    // NN
    float* dinv   = deg + NN;                              // NN
    float* h_pre  = dinv + NN;                             // NN*CHID
    float* agg    = h_pre + (long)NN * CHID;               // NN*CHID
    float* zsum   = agg + (long)NN * CHID;                 // NG*CLAT
    float* counts = zsum + NG * CLAT;                      // NG (padded to 32)
    float* xg     = counts + 32;                           // NG*CIN
    _Float16* zh  = (_Float16*)(xg + NG * CIN);            // NN*CLAT f16 (16B-aligned)

    // 1) GCN symmetric normalization degrees
    k_fill<<<NN / 256, 256, 0, stream>>>(deg, 1.0f, NN);   // self-loops
    k_deg_edge<<<NE / 256, 256, 0, stream>>>(ei, deg);
    k_dinv<<<NN / 256, 256, 0, stream>>>(deg, dinv);

    // 2) h_pre = x @ gcn_w^T   [NN, CHID]  (WMMA)
    k_gemm_abt<<<dim3(CHID / BN, NN / BM), 256, 0, stream>>>(
        x, gcn_w, nullptr, h_pre, NN, CHID, CIN, 0);

    // 3) agg = segment_sum(h_pre[row]*norm, col) + self-loop ; relu(+bias)
    k_agg_init<<<(NN * CHID) / 256, 256, 0, stream>>>(h_pre, dinv, agg);
    k_agg_edge<<<NE, CHID, 0, stream>>>(ei, dinv, h_pre, agg);
    k_bias_relu<<<(NN * CHID) / 256, 256, 0, stream>>>(agg, gcn_b);

    // 4) z_node = relu_h @ lin_w^T + lin_b   [NN, CLAT]  (WMMA)
    k_gemm_abt<<<dim3(CLAT / BN, NN / BM), 256, 0, stream>>>(
        agg, lin_w, lin_b, z_node, NN, CLAT, CHID, 0);

    // 5) global mean pool
    k_fill<<<(NG * CLAT + 255) / 256, 256, 0, stream>>>(zsum, 0.0f, NG * CLAT);
    k_fill<<<1, 256, 0, stream>>>(counts, 0.0f, NG);
    k_counts<<<NN / 256, 256, 0, stream>>>(batch, counts);
    k_pool_sum<<<(NN * CLAT) / 256, 256, 0, stream>>>(batch, z_node, zsum);
    k_pool_div<<<(NG * CLAT + 255) / 256, 256, 0, stream>>>(zsum, counts, z_graph);

    // 6) x_hat = (z_graph @ dec_w^T + dec_b)[batch]
    k_decoder<<<(NG * CIN + 255) / 256, 256, 0, stream>>>(z_graph, dec_w, dec_b, xg);
    k_gather_xhat<<<(NN * CIN) / 256, 256, 0, stream>>>(batch, xg, x_hat);

    // 7) a_hat = sigmoid(z @ z^T): f16 convert once, then async-LDS WMMA kernel
    k_z2h<<<(NN * CLAT) / 256, 256, 0, stream>>>(z_node, zh);
    k_ahat<<<dim3(NN / 64, NN / 128), 256, 0, stream>>>(zh, a_hat);
}
